// MultiHeadSelfAttention_85246510891575
// MI455X (gfx1250) — matrix operand order repaired and checked by reading, not compiled, not run on hardware
//
#include <hip/hip_runtime.h>
#include <hip/hip_bf16.h>
#include <stdint.h>

typedef __attribute__((ext_vector_type(16))) __bf16 v16bf;
typedef __attribute__((ext_vector_type(8)))  __bf16 v8bf;
typedef __attribute__((ext_vector_type(8)))  float  v8f;
typedef __attribute__((ext_vector_type(4)))  unsigned u32x4;
typedef __attribute__((ext_vector_type(8)))  int      i32x8;
typedef __attribute__((ext_vector_type(4)))  int      i32x4;

#define WMMA_BF16(a, b, c) \
  __builtin_amdgcn_wmma_f32_16x16x32_bf16(false, (a), false, (b), (short)0, (c), false, false)

// ---------------------------------------------------------------------------
// Fragment loaders (CDNA5 wave32 WMMA layouts, CDNA5 ISA §7.12.2)
__device__ __forceinline__ v16bf load_a16(const __bf16* src, int ld, int row0,
                                          int k0, int lane) {
  int m = lane & 15, h = lane >> 4;
  union { v16bf v; v8bf b[2]; } u;
  u.b[0] = *(const v8bf*)(src + (row0 + m) * ld + k0 + h * 8);
  u.b[1] = *(const v8bf*)(src + (row0 + m) * ld + k0 + 16 + h * 8);
  return u.v;
}

__device__ __forceinline__ v16bf load_bT(const __bf16* srcT, int ld, int k0,
                                         int col0, int lane) {
  int n = lane & 15, h = lane >> 4;
  union { v16bf v; v8bf b[2]; } u;
  u.b[0] = *(const v8bf*)(srcT + (col0 + n) * ld + k0 + h * 8);
  u.b[1] = *(const v8bf*)(srcT + (col0 + n) * ld + k0 + 16 + h * 8);
  return u.v;
}

// ---------------------------------------------------------------------------
// TDM: DMA a 2-D tile of bf16 (data_size=2B) from global memory into LDS.
// Descriptor packing per CDNA5 ISA (group 0) / §8.4
// (group 1).  tile rows are written contiguously to LDS (row-major fill).
__device__ __forceinline__ void tdm_load_2d(unsigned lds_off, const void* gptr,
                                            unsigned tensor_d0, unsigned tensor_d1,
                                            unsigned tile_d0, unsigned tile_d1,
                                            unsigned long long stride0) {
  unsigned long long ga = (unsigned long long)(uintptr_t)gptr;
  u32x4 g0;
  g0.x = 1u;                                            // count=1, user mode
  g0.y = lds_off;                                       // lds_addr [63:32]
  g0.z = (unsigned)ga;                                  // global_addr [95:64]
  g0.w = (unsigned)((ga >> 32) & 0x01FFFFFFu)           // global_addr [120:96]
       | 0x80000000u;                                   // type=2 ("image")
  // group1: [17:16]=data_size(1=2B)  [79:48]=tensor_dim0  [111:80]=tensor_dim1
  //         [127:112]=tile_dim0  [143:128]=tile_dim1  [207:160]=dim0_stride
  unsigned long long q0 = (1ull << 16)
                        | ((unsigned long long)(tensor_d0 & 0xFFFFu) << 48);
  unsigned long long q1 = ((unsigned long long)(tensor_d0 >> 16) & 0xFFFFull)
                        | ((unsigned long long)tensor_d1 << 16)
                        | ((unsigned long long)(tile_d0 & 0xFFFFu) << 48);
  unsigned long long q2 = (unsigned long long)(tile_d1 & 0xFFFFu)
                        | ((stride0 & 0xFFFFFFFFull) << 32);
  unsigned long long q3 = (stride0 >> 32) & 0xFFFFull;  // dim1_stride unused = 0
  union { unsigned long long q[4]; i32x8 v; } g1;
  g1.q[0] = q0; g1.q[1] = q1; g1.q[2] = q2; g1.q[3] = q3;
  i32x4 z4 = {0, 0, 0, 0};
#if __has_include(<hip/amd_detail/amd_gfx1250_TDM.h>)
  i32x8 z8 = {};
  __builtin_amdgcn_tensor_load_to_lds(g0, g1.v, z4, z4, z8, 0);
#else
  __builtin_amdgcn_tensor_load_to_lds(g0, g1.v, z4, z4, 0);
#endif
}

// ---------------------------------------------------------------------------
// Kernel 1: fp32 -> bf16 conversion.  Xt[n*256+c] = x[c*4096+n]; weights copied.
__global__ void convert_kernel(const float* __restrict__ x,
                               const float* __restrict__ wq,
                               const float* __restrict__ wk,
                               const float* __restrict__ wv,
                               const float* __restrict__ wo,
                               __bf16* __restrict__ Xt, __bf16* __restrict__ Wq,
                               __bf16* __restrict__ Wk, __bf16* __restrict__ Wv,
                               __bf16* __restrict__ Wo) {
  int tid = blockIdx.x * blockDim.x + threadIdx.x;
  if (tid < 256 * 4096) {
    int c = tid >> 12, n = tid & 4095;      // coalesced read over n
    Xt[n * 256 + c] = (__bf16)x[tid];
  } else {
    int t = tid - 256 * 4096;               // 4 * 65536 weight elements
    int which = t >> 16, i = t & 65535;
    const float* src = which == 0 ? wq : which == 1 ? wk : which == 2 ? wv : wo;
    __bf16* dst = which == 0 ? Wq : which == 1 ? Wk : which == 2 ? Wv : Wo;
    dst[i] = (__bf16)src[i];
  }
}

// ---------------------------------------------------------------------------
// Kernel 2: QKV projection. One wave per 16x16 output tile, K-loop over 256.
__global__ void proj_kernel(const __bf16* __restrict__ Xt,
                            const __bf16* __restrict__ Wq,
                            const __bf16* __restrict__ Wk,
                            const __bf16* __restrict__ Wv,
                            __bf16* __restrict__ qbuf, __bf16* __restrict__ kbuf,
                            __bf16* __restrict__ vbuf) {
  int wave = blockIdx.x * (blockDim.x >> 5) + (threadIdx.x >> 5);
  int lane = threadIdx.x & 31;
  int mat = wave >> 12;                // 0=Q 1=K 2=V  (4096 tiles each)
  int tile = wave & 4095;
  int mt = tile >> 8;                  // o-tile 0..15
  int nt = tile & 255;                 // n-tile 0..255
  const __bf16* W = (mat == 0) ? Wq : (mat == 1) ? Wk : Wv;

  v8f acc = {};
#pragma unroll
  for (int k0 = 0; k0 < 256; k0 += 32) {
    v16bf a = load_a16(W, 256, mt * 16, k0, lane);
    v16bf b = load_bT(Xt, 256, k0, nt * 16, lane);
    acc = WMMA_BF16(a, b, acc);
  }
  int half = lane >> 4, n = nt * 16 + (lane & 15);
  const float qscale = 0.17677669529663687f;  // 32^-0.5
#pragma unroll
  for (int r = 0; r < 8; ++r) {
    int o = mt * 16 + r + 8 * half;    // C-layout: M = r + 8*(lane>>4)
    float v = acc[r];
    if (mat == 2) {
      vbuf[o * 4096 + n] = (__bf16)v;
    } else {
      int h = o >> 5, j = o & 31;
      __bf16* dst = (mat == 0) ? qbuf : kbuf;
      float s = (mat == 0) ? v * qscale : v;
      dst[(h * 4096 + n) * 32 + j] = (__bf16)s;
    }
  }
}

// ---------------------------------------------------------------------------
// Kernel 3: flash attention. 8 waves/block share one head. K/V tiles are DMA'd
// into double-buffered LDS by the Tensor Data Mover (wave 0 issues, TENSORcnt
// tracks); DMA of block i+1 overlaps compute of block i. Row sums come from an
// extra WMMA against an all-ones B matrix.
__global__ void attn_kernel(const __bf16* __restrict__ qbuf,
                            const __bf16* __restrict__ kbuf,
                            const __bf16* __restrict__ vbuf,
                            __bf16* __restrict__ obuf /* Ot [4096][256] */) {
  __shared__ __align__(32) __bf16 Klds[2][32 * 32];  // [key][d]   2x2 KB
  __shared__ __align__(32) __bf16 Vlds[2][32 * 32];  // [d][key]   2x2 KB
  __shared__ __align__(32) __bf16 Plds[8][16 * 32];  // per-wave P   8 KB
  int t = threadIdx.x;
  int wave = t >> 5, lane = t & 31;
  int head = blockIdx.x >> 5;                        // 32 blocks per head
  int rb = (blockIdx.x & 31) * 128 + wave * 16;      // query-row base
  const __bf16* Qh = qbuf + head * 4096 * 32;        // [N,32]
  const __bf16* Kh = kbuf + head * 4096 * 32;        // [N,32]
  const __bf16* Vh = vbuf + head * 32 * 4096;        // [32,N]
  __bf16* P = &Plds[wave][0];

  unsigned klo[2] = { (unsigned)(uintptr_t)&Klds[0][0],
                      (unsigned)(uintptr_t)&Klds[1][0] };
  unsigned vlo[2] = { (unsigned)(uintptr_t)&Vlds[0][0],
                      (unsigned)(uintptr_t)&Vlds[1][0] };

  v16bf qa = load_a16(Qh, 32, rb, 0, lane);          // Q fragment (K-dim = 32)
  v16bf ones;
#pragma unroll
  for (int i = 0; i < 16; ++i) ones[i] = (__bf16)1.0f;

  v8f o0 = {}, o1 = {};
  float rmax[8], rsum[8], alpha[8];
#pragma unroll
  for (int r = 0; r < 8; ++r) { rmax[r] = -3.0e38f; rsum[r] = 0.0f; }

  // preload key-block 0 into buffer 0
  if (wave == 0) {
    // K tile: 32 keys x 32 d, contiguous 2048 bf16 -> 1-D tile
    tdm_load_2d(klo[0], Kh, 2048, 1, 2048, 0, 2048ull);
    // V tile: 32 d-rows x 32 keys, row stride 4096 -> 2-D tile
    tdm_load_2d(vlo[0], Vh, 4096, 32, 32, 32, 4096ull);
  }

  for (int kb = 0, ib = 0; kb < 4096; kb += 32, ++ib) {
    int cur = ib & 1, nxt = cur ^ 1;
    if (wave == 0) {
      if (kb + 32 < 4096) {
        tdm_load_2d(klo[nxt], Kh + (kb + 32) * 32, 2048, 1, 2048, 0, 2048ull);
        tdm_load_2d(vlo[nxt], Vh + (kb + 32), 4096, 32, 32, 32, 4096ull);
        __builtin_amdgcn_s_wait_tensorcnt(2);  // only next pair still in flight
      } else {
        __builtin_amdgcn_s_wait_tensorcnt(0);
      }
    }
    __syncthreads();                           // publish current buffer
    const __bf16* Kt = &Klds[cur][0];
    const __bf16* Vt = &Vlds[cur][0];

    // S = Q * K^T : B[k=d][j=key] = Kt[key*32 + d]
    v16bf b0 = load_bT(Kt, 32, 0, 0, lane);
    v16bf b1 = load_bT(Kt, 32, 0, 16, lane);
    v8f s0 = {}, s1 = {};
    s0 = WMMA_BF16(qa, b0, s0);
    s1 = WMMA_BF16(qa, b1, s1);

    int half = lane >> 4, col = lane & 15;
#pragma unroll
    for (int r = 0; r < 8; ++r) {
      float mloc = fmaxf(s0[r], s1[r]);
#pragma unroll
      for (int m = 1; m < 16; m <<= 1) mloc = fmaxf(mloc, __shfl_xor(mloc, m, 32));
      float mnew = fmaxf(rmax[r], mloc);
      alpha[r] = __expf(rmax[r] - mnew);
      rmax[r] = mnew;
      float p0 = __expf(s0[r] - mnew);
      float p1 = __expf(s1[r] - mnew);
      o0[r] *= alpha[r];
      o1[r] *= alpha[r];
      int row = r + 8 * half;                  // C-layout -> LDS [16][32]
      P[row * 32 + col]      = (__bf16)p0;
      P[row * 32 + 16 + col] = (__bf16)p1;
    }
    // re-read P in A-layout (DS ops are in-order within a wave)
    v16bf pa = load_a16(P, 32, 0, 0, lane);

    // row sums via WMMA: D = P * ones -> row sum replicated across lanes
    v8f sb = {};
    sb = WMMA_BF16(pa, ones, sb);
#pragma unroll
    for (int r = 0; r < 8; ++r) rsum[r] = rsum[r] * alpha[r] + sb[r];

    // O += P * V : B[k=key][j=d] = Vt[d*32 + k]
    v16bf v0 = load_bT(Vt, 32, 0, 0, lane);
    v16bf v1 = load_bT(Vt, 32, 0, 16, lane);
    o0 = WMMA_BF16(pa, v0, o0);
    o1 = WMMA_BF16(pa, v1, o1);
    __syncthreads();   // all waves done with 'cur' before TDM overwrites it
  }

  int half = lane >> 4, col = lane & 15;
#pragma unroll
  for (int r = 0; r < 8; ++r) {
    int n = rb + r + 8 * half;
    float inv = 1.0f / rsum[r];
    obuf[n * 256 + head * 32 + col]      = (__bf16)(o0[r] * inv);
    obuf[n * 256 + head * 32 + 16 + col] = (__bf16)(o1[r] * inv);
  }
}

// ---------------------------------------------------------------------------
// Kernel 4: output projection (wo * O) fused with depthwise 3x3 conv + add.
__global__ void outproj_kernel(const __bf16* __restrict__ Wo,
                               const __bf16* __restrict__ Ot,
                               const float* __restrict__ x,
                               const float* __restrict__ wl,
                               float* __restrict__ out) {
  int wave = blockIdx.x * (blockDim.x >> 5) + (threadIdx.x >> 5);
  int lane = threadIdx.x & 31;
  int mt = wave >> 8;                  // c-tile 0..15
  int nt = wave & 255;                 // n-tile 0..255

  v8f acc = {};
#pragma unroll
  for (int k0 = 0; k0 < 256; k0 += 32) {
    v16bf a = load_a16(Wo, 256, mt * 16, k0, lane);
    v16bf b = load_bT(Ot, 256, k0, nt * 16, lane);
    acc = WMMA_BF16(a, b, acc);
  }
  int half = lane >> 4, n = nt * 16 + (lane & 15);
  int hh = n >> 6, ww = n & 63;
#pragma unroll
  for (int r = 0; r < 8; ++r) {
    int c = mt * 16 + r + 8 * half;
    const float* xi = x + c * 4096;
    const float* wc = wl + c * 9;
    float conv = 0.0f;
#pragma unroll
    for (int dh = -1; dh <= 1; ++dh) {
      int hy = hh + dh;
      if (hy < 0 || hy > 63) continue;
#pragma unroll
      for (int dw = -1; dw <= 1; ++dw) {
        int wx = ww + dw;
        if (wx < 0 || wx > 63) continue;
        conv += xi[hy * 64 + wx] * wc[(dh + 1) * 3 + (dw + 1)];
      }
    }
    out[c * 4096 + n] = acc[r] + conv;
  }
}

// ---------------------------------------------------------------------------
extern "C" void kernel_launch(void* const* d_in, const int* in_sizes, int n_in,
                              void* d_out, int out_size, void* d_ws, size_t ws_size,
                              hipStream_t stream) {
  const float* x  = (const float*)d_in[0];
  const float* wq = (const float*)d_in[1];
  const float* wk = (const float*)d_in[2];
  const float* wv = (const float*)d_in[3];
  const float* wo = (const float*)d_in[4];
  const float* wl = (const float*)d_in[5];
  float* out = (float*)d_out;

  char* ws = (char*)d_ws;
  __bf16* Xt = (__bf16*)ws;                         // [4096][256]  2 MiB
  __bf16* Wq = (__bf16*)(ws + (2u << 20));          // [256][256]   128 KiB each
  __bf16* Wk = Wq + 65536;
  __bf16* Wv = Wk + 65536;
  __bf16* Wo = Wv + 65536;
  __bf16* qbuf = (__bf16*)(ws + (2u << 20) + 4u * 65536u * 2u);  // [8][4096][32]
  __bf16* kbuf = qbuf + 4096 * 256;                 // [8][4096][32]
  __bf16* vbuf = kbuf + 4096 * 256;                 // [256][4096]
  __bf16* obuf = vbuf + 4096 * 256;                 // Ot [4096][256]

  convert_kernel<<<5120, 256, 0, stream>>>(x, wq, wk, wv, wo, Xt, Wq, Wk, Wv, Wo);
  proj_kernel<<<1536, 256, 0, stream>>>(Xt, Wq, Wk, Wv, qbuf, kbuf, vbuf);
  attn_kernel<<<256, 256, 0, stream>>>(qbuf, kbuf, vbuf, obuf);
  outproj_kernel<<<512, 256, 0, stream>>>(Wo, obuf, x, wl, out);
}
